// TemporalBiasMultiHeadSelfAttention_88484916232434
// MI455X (gfx1250) — matrix operand order repaired and checked by reading, not compiled, not run on hardware
//
#include <hip/hip_runtime.h>
#include <hip/hip_bf16.h>
#include <math.h>

// ---------------------------------------------------------------------------
// Types for CDNA5 WMMA (wave32). bf16 operands, f32 accumulate.
// ---------------------------------------------------------------------------
typedef __bf16 bf16_t;
typedef __attribute__((ext_vector_type(16))) __bf16 v16bf;
typedef __attribute__((ext_vector_type(8)))  __bf16 v8bf;
typedef __attribute__((ext_vector_type(4)))  __bf16 v4bf;
typedef __attribute__((ext_vector_type(8)))  float  v8f;

#define WMMA_BF16(a, b, c) \
  __builtin_amdgcn_wmma_f32_16x16x32_bf16(false, (a), false, (b), (short)0, (c), false, false)

// Problem constants (match reference)
#define B   2
#define S   2048
#define D   1024
#define H   16
#define HD  64
#define ROWS (B * S)          // 4096
#define N_QKV (3 * D)         // 3072

// ---------------------------------------------------------------------------
// Fragment loaders (this code's contiguous-K order; A and B use the same one).
// A (16x32 bf16): lane l -> m = l%16, khalf = l/16.
//   elements 0..15 = 16 contiguous K starting at khalf*16
// B (32x16 bf16) loaded from BT[N][K] (row-major over K):
//   lane l -> n = l%16, elements = 16 contiguous K starting at (l/16)*16.
// C/D (16x16 f32): lane l, vgpr r -> row m = r + 8*(l/16), col n = l%16.
// ---------------------------------------------------------------------------
static __device__ __forceinline__ v16bf load_afrag(const bf16_t* __restrict__ a,
                                                   int ldk, int row, int kk, int khalf) {
  const bf16_t* p = a + (long)row * ldk + kk + khalf * 16;
  v8bf c0 = *(const v8bf*)p;
  v8bf c1 = *(const v8bf*)(p + 8);
  v16bf r;
#pragma unroll
  for (int i = 0; i < 8; ++i) { r[i] = c0[i]; r[i + 8] = c1[i]; }
  return r;
}

static __device__ __forceinline__ v16bf load_bfrag(const bf16_t* __restrict__ bt,
                                                   long ldk, int n, int kbase) {
  return *(const v16bf*)(bt + (long)n * ldk + kbase);
}

// ---------------------------------------------------------------------------
// Weight prep: w [K][N] f32  ->  wt [N][K] bf16 (transposed for B fragments)
// ---------------------------------------------------------------------------
__global__ void k_prep_w(const float* __restrict__ w, bf16_t* __restrict__ wt,
                         int N, int K) {
  long t = (long)blockIdx.x * 256 + threadIdx.x;
  long total = (long)N * K;
  if (t >= total) return;
  int n = (int)(t / K), k = (int)(t % K);
  wt[t] = (bf16_t)w[(long)k * N + n];
}

// bias table: tab[i] = beta * log1p(i)
__global__ void k_tab(const float* __restrict__ beta, float* __restrict__ tab) {
  int i = blockIdx.x * 256 + threadIdx.x;
  if (i < S) tab[i] = beta[0] * log1pf((float)i);
}

// ---------------------------------------------------------------------------
// LayerNorm: x [ROWS][D] f32 -> xn bf16. One block per row (256 thr, 4/thread)
// ---------------------------------------------------------------------------
__global__ __launch_bounds__(256) void k_ln(const float* __restrict__ x,
                                            const float* __restrict__ gamma,
                                            const float* __restrict__ lnb,
                                            bf16_t* __restrict__ xn) {
  long row = blockIdx.x;
  const float* xr = x + row * D;
  int t = threadIdx.x;
  float4 v = ((const float4*)xr)[t];
  float s  = v.x + v.y + v.z + v.w;
  float s2 = v.x * v.x + v.y * v.y + v.z * v.z + v.w * v.w;
#pragma unroll
  for (int off = 1; off < 32; off <<= 1) {
    s  += __shfl_xor(s,  off, 32);
    s2 += __shfl_xor(s2, off, 32);
  }
  __shared__ float rs[8], rs2[8];
  int w = t >> 5, lane = t & 31;
  if (lane == 0) { rs[w] = s; rs2[w] = s2; }
  __syncthreads();
  if (w == 0) {
    float a = lane < 8 ? rs[lane]  : 0.f;
    float b = lane < 8 ? rs2[lane] : 0.f;
#pragma unroll
    for (int off = 1; off < 8; off <<= 1) {
      a += __shfl_xor(a, off, 32);
      b += __shfl_xor(b, off, 32);
    }
    if (lane == 0) { rs[0] = a; rs2[0] = b; }
  }
  __syncthreads();
  float mean = rs[0] * (1.f / D);
  float var  = rs2[0] * (1.f / D) - mean * mean;
  float rstd = rsqrtf(var + 1e-5f);
  float4 g  = ((const float4*)gamma)[t];
  float4 bb = ((const float4*)lnb)[t];
  v4bf o;
  o[0] = (bf16_t)((v.x - mean) * rstd * g.x + bb.x);
  o[1] = (bf16_t)((v.y - mean) * rstd * g.y + bb.y);
  o[2] = (bf16_t)((v.z - mean) * rstd * g.z + bb.z);
  o[3] = (bf16_t)((v.w - mean) * rstd * g.w + bb.w);
  *(v4bf*)(xn + row * D + t * 4) = o;
}

// ---------------------------------------------------------------------------
// GEMM: C[m][n] = sum_k A[m][k] * BT[n][k], wave tile 16x64, block 8 waves
// EPI 0: QKV epilogue (+b_qkv, scale Q, scatter to Q/K/Vt bf16)
// EPI 1: out-proj epilogue (+b_out + residual, f32 out)
// ---------------------------------------------------------------------------
template <int EPI>
__global__ __launch_bounds__(256) void k_gemm(
    const bf16_t* __restrict__ A, const bf16_t* __restrict__ BT, int K,
    const float* __restrict__ bias, const float* __restrict__ resid,
    float* __restrict__ outf, bf16_t* __restrict__ qb,
    bf16_t* __restrict__ kbuf, bf16_t* __restrict__ vtb) {
  int lane = threadIdx.x & 31, w = threadIdx.x >> 5;
  int khalf = lane >> 4, ncol = lane & 15;
  int m0 = blockIdx.y * 128 + w * 16;
  int n0 = blockIdx.x * 64;
  int mrow = m0 + ncol;  // A-fragment row for this lane (m = lane % 16)

  v8f c[4] = {v8f{0.f}, v8f{0.f}, v8f{0.f}, v8f{0.f}};
  for (int kk = 0; kk < K; kk += 32) {
    // keep the 23.3 TB/s stream ahead of the WMMA pipe (global_prefetch_b8)
    if (kk + 32 < K) {
      __builtin_prefetch(A + (long)mrow * K + kk + 32, 0, 3);
      __builtin_prefetch(BT + (long)(n0 + lane) * K + kk + 32, 0, 3);
    }
    v16bf a = load_afrag(A, K, mrow, kk, khalf);
    int kbase = kk + khalf * 16;
#pragma unroll
    for (int f = 0; f < 4; ++f) {
      v16bf b = load_bfrag(BT, K, n0 + f * 16 + ncol, kbase);
      c[f] = WMMA_BF16(a, b, c[f]);
    }
  }

  if constexpr (EPI == 0) {
    int part = n0 / D;             // 0=Q 1=K 2=V (tiles never cross: 64|1024)
    int head = (n0 % D) / HD;      // 64-wide wave tile == one head
#pragma unroll
    for (int f = 0; f < 4; ++f) {
      int hdc = f * 16 + ncol;     // n0 % 64 == 0
      float bia = bias[n0 + f * 16 + ncol];
#pragma unroll
      for (int r = 0; r < 8; ++r) {
        int g = m0 + r + 8 * khalf;
        int bi = g >> 11, srow = g & (S - 1);
        float v = c[f][r] + bia;
        long bh = (long)(bi * H + head);
        if (part == 0) {
          qb[(bh * S + srow) * HD + hdc] = (bf16_t)(v * 0.125f);  // 1/sqrt(64)
        } else if (part == 1) {
          kbuf[(bh * S + srow) * HD + hdc] = (bf16_t)v;
        } else {
          vtb[(bh * HD + hdc) * S + srow] = (bf16_t)v;            // V transposed
        }
      }
    }
  } else {
#pragma unroll
    for (int f = 0; f < 4; ++f) {
      int n = n0 + f * 16 + ncol;
      float bia = bias[n];
#pragma unroll
      for (int r = 0; r < 8; ++r) {
        long g = m0 + r + 8 * khalf;
        outf[g * D + n] = c[f][r] + bia + resid[g * D + n];
      }
    }
  }
}

// ---------------------------------------------------------------------------
// Flash attention with temporal log bias. Block = 8 waves, each wave owns a
// 16-query strip; keys swept in blocks of 32. Bias via LDS table lookup.
// Row-sum of P is offloaded to the matrix pipe (P @ ones) so the only
// cross-lane VALU reduction left is the running max.
// ---------------------------------------------------------------------------
__global__ __launch_bounds__(256) void k_attn(const bf16_t* __restrict__ qb,
                                              const bf16_t* __restrict__ kb,
                                              const bf16_t* __restrict__ vtb,
                                              const float* __restrict__ tab,
                                              bf16_t* __restrict__ ctx) {
  __shared__ float  tab_s[S];
  __shared__ bf16_t pst[8][16][32];  // per-wave P staging (C-layout -> A-layout)
  int tid = threadIdx.x;
#pragma unroll
  for (int i = 0; i < S / 256; ++i) tab_s[tid + i * 256] = tab[tid + i * 256];
  __syncthreads();

  int lane = tid & 31, w = tid >> 5;
  int khalf = lane >> 4, nc = lane & 15;
  int bh = blockIdx.y;
  int q0 = blockIdx.x * 128 + w * 16;
  const bf16_t* qp = qb  + (long)bh * S * HD;
  const bf16_t* kp = kb  + (long)bh * S * HD;
  const bf16_t* vp = vtb + (long)bh * HD * S;

  v16bf aq[2];
#pragma unroll
  for (int f = 0; f < 2; ++f) aq[f] = load_afrag(qp, HD, q0 + nc, f * 32, khalf);

  v16bf ones;
#pragma unroll
  for (int i = 0; i < 16; ++i) ones[i] = (bf16_t)1.0f;

  float mrun[8];
#pragma unroll
  for (int r = 0; r < 8; ++r) mrun[r] = -INFINITY;
  v8f o[4] = {v8f{0.f}, v8f{0.f}, v8f{0.f}, v8f{0.f}};
  v8f lf = v8f{0.f};  // running row-sum of P (every column identical)

  for (int kbk = 0; kbk < S; kbk += 32) {
    // prefetch next key block's K rows and V columns
    if (kbk + 32 < S) {
      __builtin_prefetch(kp + (long)(kbk + 32 + lane) * HD, 0, 3);
      __builtin_prefetch(vp + (long)(lane & 15) * 4 * S + kbk + 32, 0, 3);
    }
    // --- scores: S0 (keys kbk..+15), S1 (keys kbk+16..+31), contract hd=64
    v8f s0 = v8f{0.f}, s1 = v8f{0.f};
#pragma unroll
    for (int f = 0; f < 2; ++f) {
      v16bf b0 = load_bfrag(kp, HD, kbk + nc,      f * 32 + khalf * 16);
      v16bf b1 = load_bfrag(kp, HD, kbk + 16 + nc, f * 32 + khalf * 16);
      s0 = WMMA_BF16(aq[f], b0, s0);
      s1 = WMMA_BF16(aq[f], b1, s1);
    }
    // --- bias + online softmax (row m = r + 8*khalf, col n = nc)
    float alp[8];
#pragma unroll
    for (int r = 0; r < 8; ++r) {
      int qi = q0 + r + 8 * khalf;
      int d0 = qi - (kbk + nc);      d0 = d0 < 0 ? -d0 : d0;
      int d1 = qi - (kbk + 16 + nc); d1 = d1 < 0 ? -d1 : d1;
      float v0 = s0[r] + tab_s[d0];
      float v1 = s1[r] + tab_s[d1];
      float vm = fmaxf(v0, v1);
#pragma unroll
      for (int off = 1; off < 16; off <<= 1) vm = fmaxf(vm, __shfl_xor(vm, off, 16));
      float mnew = fmaxf(mrun[r], vm);
      float al = __expf(mrun[r] - mnew);
      float p0 = __expf(v0 - mnew);
      float p1 = __expf(v1 - mnew);
      mrun[r] = mnew;
      alp[r] = al;
      int m = r + 8 * khalf;
      pst[w][m][nc]      = (bf16_t)p0;
      pst[w][m][16 + nc] = (bf16_t)p1;
    }
    // --- rescale accumulators (O and row-sum fragment)
#pragma unroll
    for (int f = 0; f < 4; ++f)
#pragma unroll
      for (int r = 0; r < 8; ++r) o[f][r] *= alp[r];
#pragma unroll
    for (int r = 0; r < 8; ++r) lf[r] *= alp[r];
    // --- P (A-layout from LDS); rowsum via P @ ones; O += P @ V
    v16bf pa = load_afrag(&pst[w][0][0], 32, nc, 0, khalf);
    lf = WMMA_BF16(pa, ones, lf);
#pragma unroll
    for (int f = 0; f < 4; ++f) {
      v16bf bv = load_bfrag(vp, S, f * 16 + nc, kbk + khalf * 16);
      o[f] = WMMA_BF16(pa, bv, o[f]);
    }
  }

  // --- normalize and scatter ctx [b, s, h*hd] bf16
  int b = bh >> 4, h = bh & 15;
#pragma unroll
  for (int r = 0; r < 8; ++r) {
    float inv = 1.0f / lf[r];
    long g = (long)b * S + q0 + r + 8 * khalf;
#pragma unroll
    for (int f = 0; f < 4; ++f)
      ctx[g * D + h * HD + f * 16 + nc] = (bf16_t)(o[f][r] * inv);
  }
}

// ---------------------------------------------------------------------------
// Host launch
// ---------------------------------------------------------------------------
extern "C" void kernel_launch(void* const* d_in, const int* in_sizes, int n_in,
                              void* d_out, int out_size, void* d_ws, size_t ws_size,
                              hipStream_t stream) {
  const float* x      = (const float*)d_in[0];
  const float* w_qkv  = (const float*)d_in[1];
  const float* b_qkv  = (const float*)d_in[2];
  const float* w_out  = (const float*)d_in[3];
  const float* b_out  = (const float*)d_in[4];
  const float* gamma  = (const float*)d_in[5];
  const float* ln_b   = (const float*)d_in[6];
  const float* beta   = (const float*)d_in[7];
  float* out = (float*)d_out;

  // workspace carve-up (256B aligned)
  char* ws = (char*)d_ws;
  size_t off = 0;
  auto carve = [&](size_t bytes) {
    char* p = ws + off;
    off = (off + bytes + 255) & ~(size_t)255;
    return p;
  };
  bf16_t* wqkv_t = (bf16_t*)carve((size_t)N_QKV * D * 2);   // [3072][1024]
  bf16_t* wout_t = (bf16_t*)carve((size_t)D * D * 2);       // [1024][1024]
  bf16_t* xn     = (bf16_t*)carve((size_t)ROWS * D * 2);
  bf16_t* qb     = (bf16_t*)carve((size_t)B * H * S * HD * 2);
  bf16_t* kbuf   = (bf16_t*)carve((size_t)B * H * S * HD * 2);
  bf16_t* vtb    = (bf16_t*)carve((size_t)B * H * HD * S * 2);
  bf16_t* ctx    = (bf16_t*)carve((size_t)ROWS * D * 2);
  float*  tab    = (float*)carve((size_t)S * 4);
  (void)ws_size; (void)in_sizes; (void)n_in; (void)out_size;

  k_prep_w<<<(N_QKV * D + 255) / 256, 256, 0, stream>>>(w_qkv, wqkv_t, N_QKV, D);
  k_prep_w<<<(D * D + 255) / 256, 256, 0, stream>>>(w_out, wout_t, D, D);
  k_tab<<<(S + 255) / 256, 256, 0, stream>>>(beta, tab);
  k_ln<<<ROWS, 256, 0, stream>>>(x, gamma, ln_b, xn);
  k_gemm<0><<<dim3(N_QKV / 64, ROWS / 128), 256, 0, stream>>>(
      xn, wqkv_t, D, b_qkv, nullptr, nullptr, qb, kbuf, vtb);
  k_attn<<<dim3(S / 128, B * H), 256, 0, stream>>>(qb, kbuf, vtb, tab, ctx);
  k_gemm<1><<<dim3(D / 64, ROWS / 128), 256, 0, stream>>>(
      ctx, wout_t, D, b_out, x, out, nullptr, nullptr, nullptr);
}
